// conv2d_30889404793383
// MI455X (gfx1250) — matrix operand order repaired and checked by reading, not compiled, not run on hardware
//
#include <hip/hip_runtime.h>

typedef __attribute__((ext_vector_type(16))) _Float16 v16h;
typedef __attribute__((ext_vector_type(8)))  float    v8f;

#define WN 128          // output channels
#define CIN 64
#define HW 56
#define CH_STRIDE (HW * HW)   // 3136 floats per input channel plane
#define KGEMM 576       // 9 taps * 64 channels, K-order: k = rs*64 + c
#define BLOCK_M 128
#define B_STRIDE 584    // padded halfs per B row (1168B -> conflict-free b128 reads)
#define C_STRIDE 133    // padded floats per C row (conflict-free transpose reads)
#define LDS_TOTAL (WN * B_STRIDE * 2)   // 149504 B: weights resident; C reuses it

// ---------------- Kernel 1: scale = max|w| / 127 ----------------
__global__ __launch_bounds__(256) void maxabs_kernel(const float* __restrict__ w,
                                                     int n, float* __restrict__ scale_out) {
    __shared__ float red[256];
    float m = 0.0f;
    for (int i = threadIdx.x; i < n; i += 256) m = fmaxf(m, fabsf(w[i]));
    red[threadIdx.x] = m;
    __syncthreads();
    for (int s = 128; s > 0; s >>= 1) {
        if (threadIdx.x < (unsigned)s) red[threadIdx.x] = fmaxf(red[threadIdx.x], red[threadIdx.x + s]);
        __syncthreads();
    }
    if (threadIdx.x == 0) scale_out[0] = red[0] / 127.0f;
}

// --- Kernel 2: Bq[o][rs*64 + c] = (f16) round(w[o][c][r][s] / scale) (exact ints) ---
__global__ __launch_bounds__(256) void quant_kernel(const float* __restrict__ w,
                                                    const float* __restrict__ scale_p,
                                                    _Float16* __restrict__ Bq, int n) {
    int i = blockIdx.x * 256 + threadIdx.x;      // output index in k'-order
    if (i < n) {
        int o  = i / KGEMM;
        int t  = i - o * KGEMM;
        int rs = t >> 6;                         // filter tap 0..8
        int c  = t & 63;                         // input channel
        float s = scale_p[0];
        Bq[i] = (_Float16)rintf(w[(o * CIN + c) * 9 + rs] / s);  // round-half-even
    }
}

// ---------------- Kernel 3: implicit-GEMM conv via WMMA ----------------
__global__ __launch_bounds__(256) void conv_wmma_kernel(const float* __restrict__ x,
                                                        const _Float16* __restrict__ Bq,
                                                        const float* __restrict__ scale_p,
                                                        float* __restrict__ out) {
    extern __shared__ char smem[];
    _Float16* Bs = (_Float16*)smem;      // [128][B_STRIDE] quantized weights
    float*    Cs = (float*)smem;         // [128][C_STRIDE] epilogue (reuses Bs)

    const int tid  = threadIdx.x;
    const int wave = tid >> 5;
    const int lane = tid & 31;
    const int m0   = blockIdx.x * BLOCK_M;

    // Stage k' 16j..16j+7 (one int4 per 16 halfs) of each weight row into LDS as [n][k'] (row-major, padded).
    for (int idx = tid; idx < 4608; idx += 256) {
        int n  = idx / 36;
        int kc = (idx - n * 36) * 16;
        int4 v = *(const int4*)(Bq + n * KGEMM + kc);
        *(int4*)(Bs + n * B_STRIDE + kc) = v;
    }

    const float scale = scale_p[0];
    v8f acc[8] = {};                     // 8 x (16 channels) f32 accumulators

    // Per-lane A addressing (ISA A-fragment layout: lane = M row, half by lane>=16)
    const int a_m   = lane & 15;
    const int khalf = lane >> 4;         // 0: K 0-7/16-23, 1: K 8-15/24-31
    int m  = m0 + wave * 16 + a_m;
    int ow = m % HW;  int t_ = m / HW;
    int oh = t_ % HW; int nb = t_ / HW;
    // base channel of this lane's run within a tap:
    const float* xb = x + (size_t)nb * CIN * CH_STRIDE + (size_t)(khalf * 8) * CH_STRIDE;

    // Per-lane B addressing (lane = N column, K 8h..8h+7 and 16+8h..16+8h+7 per lane)
    const int b_n    = lane & 15;
    const int b_koff = khalf * 8;
    const _Float16* brow = Bs + b_n * B_STRIDE + b_koff;

    // Gather one tap (64 channels) -> 16 packed half2 words per lane.
    // Word i = pack(c-offset pair) matching v16h element pairs (2i, 2i+1).
    auto gather = [&](int rs, unsigned* hw) {
        int r  = rs / 3;
        int s  = rs - r * 3;
        int ih = oh + r - 1;
        int iw = ow + s - 1;
        bool valid = ((unsigned)ih < (unsigned)HW) && ((unsigned)iw < (unsigned)HW);
        const float* p = xb + ih * HW + iw;          // one base, const offsets
        if (valid) {
            float fv[32];
            #pragma unroll
            for (int e = 0; e < 8; ++e) {
                fv[e]      = p[(e)      * CH_STRIDE]; // step0 slots 0-7
                fv[e + 8]  = p[(e + 16) * CH_STRIDE]; // step0 slots 8-15
                fv[e + 16] = p[(e + 32) * CH_STRIDE]; // step1 slots 0-7
                fv[e + 24] = p[(e + 48) * CH_STRIDE]; // step1 slots 8-15
            }
            #pragma unroll
            for (int i = 0; i < 16; ++i) {
                auto pk = __builtin_amdgcn_cvt_pkrtz(fv[2 * i], fv[2 * i + 1]);
                hw[i] = __builtin_bit_cast(unsigned, pk);
            }
        } else {
            #pragma unroll
            for (int i = 0; i < 16; ++i) hw[i] = 0u;
        }
    };

    // Two 16x16x32 K-steps (16 WMMAs) for one tap from packed registers.
    auto compute = [&](int rs, const unsigned* hw) {
        union { v16h v; unsigned u[8]; } A0, A1;
        #pragma unroll
        for (int i = 0; i < 8; ++i) { A0.u[i] = hw[i]; A1.u[i] = hw[i + 8]; }
        const int kk = rs * 64;
        #pragma unroll
        for (int nt = 0; nt < 8; ++nt) {
            const _Float16* bp = brow + nt * 16 * B_STRIDE + kk;
            v16h bfrag;
            ((int4*)&bfrag)[0] = *(const int4*)(bp);
            ((int4*)&bfrag)[1] = *(const int4*)(bp + 16);
            acc[nt] = __builtin_amdgcn_wmma_f32_16x16x32_f16(
                false, A0.v, false, bfrag, (short)0, acc[nt], false, false);
        }
        #pragma unroll
        for (int nt = 0; nt < 8; ++nt) {
            const _Float16* bp = brow + nt * 16 * B_STRIDE + kk + 32;
            v16h bfrag;
            ((int4*)&bfrag)[0] = *(const int4*)(bp);
            ((int4*)&bfrag)[1] = *(const int4*)(bp + 16);
            acc[nt] = __builtin_amdgcn_wmma_f32_16x16x32_f16(
                false, A1.v, false, bfrag, (short)0, acc[nt], false, false);
        }
    };

    __syncthreads();                     // weights resident; no barriers in K loop

    // Software-pipelined tap loop: loads for tap t+1 issue before WMMAs of tap t.
    unsigned cur[16], nxt[16];
    gather(0, cur);
    #pragma unroll 1
    for (int rs = 0; rs < 8; rs += 2) {
        gather(rs + 1, nxt);
        compute(rs, cur);
        gather(rs + 2, cur);             // rs+2 <= 8, always a real tap
        compute(rs + 1, nxt);
    }
    compute(8, cur);

    __syncthreads();   // all waves finished reading Bs; safe to reuse as Cs

    // ---- epilogue: scale, transpose through LDS, coalesced NT stores ----
    const int mbase = wave * 16 + ((lane >> 4) << 3);   // + j -> local M row
    const int ncol  = lane & 15;
    #pragma unroll
    for (int nt = 0; nt < 8; ++nt) {
        #pragma unroll
        for (int j = 0; j < 8; ++j) {
            Cs[(mbase + j) * C_STRIDE + nt * 16 + ncol] = acc[nt][j] * scale;
        }
    }
    __syncthreads();

    for (int idx = tid; idx < BLOCK_M * WN; idx += 256) {
        int ch  = idx >> 7;          // output channel
        int row = idx & 127;         // local M row -> consecutive ow: coalesced
        int mm  = m0 + row;
        int ow2 = mm % HW;  int t2 = mm / HW;
        int oh2 = t2 % HW;  int nb2 = t2 / HW;
        float* dst = out + (((size_t)nb2 * WN + ch) * HW + oh2) * HW + ow2;
        __builtin_nontemporal_store(Cs[row * C_STRIDE + ch], dst);  // stream 51MB out
    }
}

extern "C" void kernel_launch(void* const* d_in, const int* in_sizes, int n_in,
                              void* d_out, int out_size, void* d_ws, size_t ws_size,
                              hipStream_t stream) {
    const float* x = (const float*)d_in[0];        // [32,64,56,56]
    const float* w = (const float*)d_in[1];        // [128,64,3,3]
    float* out = (float*)d_out;                    // [32,128,56,56]

    char* ws = (char*)d_ws;
    float*    scale = (float*)ws;                  // 4 B
    _Float16* Bq    = (_Float16*)(ws + 16);        // 128*576 f16 = 147456 B

    const int wn = WN * KGEMM;                     // 73728 weight elements
    maxabs_kernel<<<1, 256, 0, stream>>>(w, wn, scale);
    quant_kernel<<<(wn + 255) / 256, 256, 0, stream>>>(w, scale, Bq, wn);

    const int grid_m = (32 * HW * HW) / BLOCK_M;   // 100352 / 128 = 784
    conv_wmma_kernel<<<grid_m, 256, LDS_TOTAL, stream>>>(x, Bq, scale, out);
}
